// StoMajorityVoteDiff_78503412236666
// MI455X (gfx1250) — matrix operand order repaired and checked by reading, not compiled, not run on hardware
//
#include <hip/hip_runtime.h>
#include <hip/hip_bf16.h>
#include <math.h>

// ---------------------------------------------------------------------------
// StoMajorityVoteDiff on MI455X (gfx1250, wave32).
//
//   k_setup (1 block): post = exp(post_log); S; mv = post/S (Dirichlet mean;
//           JAX threefry draw not reproducible); KL scalar; 256-pt Gauss-
//           Legendre nodes via f64 Newton; nodes packed as float4
//           (logt, log1mt, logw, 0); WMMA B-fragment pack (f16 hi/lo split,
//           columns: 0=post_hi 1=post_lo 2=mv_hi 3=mv_lo).
//   k_main  (B/128 blocks, 8 waves): per wave one 16-row tile of `out`,
//           16x v_wmma_f32_16x16x32_f16 (+-1 exact in f16, hi+lo columns give
//           fp32-accurate dots), then single-pass log-space quadrature with
//           an analytic max bound (no max pass). Node table staged into LDS
//           with global_load_async_to_lds_b128 (ASYNCcnt path).
//
// HBM-bound: 128 MB of `out` @ 23.3 TB/s ~ 5.5 us floor.
// ---------------------------------------------------------------------------

typedef __attribute__((ext_vector_type(16))) _Float16 v16h;
typedef __attribute__((ext_vector_type(8)))  float    v8f;

#define NV 512     // voters (V)
#define NQ 256     // quadrature nodes
// workspace float offsets
#define OFF_S      0
#define OFF_KL     1
#define OFF_NODES  8                    // float4[NQ] -> 1024 floats, 16B aligned
#define OFF_BPACK  (8 + 4 * NQ)         // float offset 1032 -> byte 4128 (32B aligned)

__device__ __forceinline__ float digammaf_dev(float x) {
  float r = 0.0f;
  while (x < 6.0f) { r -= 1.0f / x; x += 1.0f; }
  float inv = 1.0f / x, inv2 = inv * inv;
  return r + logf(x) - 0.5f * inv
           - inv2 * (1.0f / 12.0f - inv2 * (1.0f / 120.0f - inv2 * (1.0f / 252.0f)));
}

// Evaluate P_NQ(x) and P'_NQ(x) via recurrence (f64).
__device__ __forceinline__ void legendre_eval(double x, double* p, double* dp) {
  double p0 = 1.0, p1 = x;
  for (int k = 2; k <= NQ; ++k) {
    double p2 = ((2.0 * k - 1.0) * x * p1 - (k - 1.0) * p0) / (double)k;
    p0 = p1; p1 = p2;
  }
  *p  = p1;
  *dp = (double)NQ * (x * p1 - p0) / (x * x - 1.0);
}

__global__ __launch_bounds__(512) void k_setup(const float* __restrict__ post_log,
                                               const float* __restrict__ prior,
                                               float* __restrict__ ws) {
  __shared__ float red[512];
  __shared__ float s_post[NV];
  __shared__ float s_mv[NV];
  const int t = threadIdx.x;

  float pv = expf(post_log[t]);
  float pr = prior[t];
  s_post[t] = pv;
  red[t] = pv;
  __syncthreads();
  for (int off = 256; off > 0; off >>= 1) {
    if (t < off) red[t] += red[t + off];
    __syncthreads();
  }
  const float S = red[0];
  __syncthreads();

  s_mv[t] = pv / S;

  red[t] = pr;
  __syncthreads();
  for (int off = 256; off > 0; off >>= 1) {
    if (t < off) red[t] += red[t + off];
    __syncthreads();
  }
  const float Sp = red[0];
  __syncthreads();

  const float digS = digammaf_dev(S);
  red[t] = -lgammaf(pv) + lgammaf(pr) + (pv - pr) * (digammaf_dev(pv) - digS);
  __syncthreads();
  for (int off = 256; off > 0; off >>= 1) {
    if (t < off) red[t] += red[t + off];
    __syncthreads();
  }
  if (t == 0) {
    ws[OFF_S]  = S;
    ws[OFF_KL] = lgammaf(S) - lgammaf(Sp) + red[0];
  }

  // Gauss-Legendre nodes/weights on [-1,1] -> mapped to (0, 0.5); pack float4.
  if (t < NQ) {
    double x = cos(3.14159265358979323846 * ((double)t + 0.75) / ((double)NQ + 0.5));
    double p, dp;
    for (int it = 0; it < 8; ++it) {
      legendre_eval(x, &p, &dp);
      x -= p / dp;
    }
    legendre_eval(x, &p, &dp);
    double w  = 2.0 / ((1.0 - x * x) * dp * dp);
    double tn = 0.25 * (x + 1.0);
    float4* np = (float4*)(ws + OFF_NODES);
    np[t] = make_float4((float)log(tn), (float)log1p(-tn), (float)log(0.25 * w), 0.0f);
  }
  __syncthreads();

  // WMMA B fragment pack: 16 k-chunks x 32 lanes x 16 halves.
  // B element (k, n): fragment lane = 16*((k%32)/16) + n, half = k%16.
  // Columns: 0 = hi(post), 1 = lo(post), 2 = hi(mv), 3 = lo(mv), rest 0.
  _Float16* bp = (_Float16*)(ws + OFF_BPACK);
  for (int idx = t; idx < 16 * 32 * 16; idx += 512) {
    const int c    = idx >> 9;
    const int rem  = idx & 511;
    const int lane = rem >> 4;
    const int h    = rem & 15;
    const int k    = c * 32 + (lane >> 4) * 16 + h;
    const int n    = lane & 15;
    _Float16 vh = (_Float16)0.0f;
    if (n < 4) {
      const float val = (n < 2) ? s_post[k] : s_mv[k];
      const _Float16 hi = (_Float16)val;
      vh = (n & 1) ? (_Float16)(val - (float)hi) : hi;
    }
    bp[idx] = vh;
  }
}

__global__ __launch_bounds__(256) void k_main(const int* __restrict__ outM,
                                              const int* __restrict__ y,
                                              const float* __restrict__ ws,
                                              float* __restrict__ dst) {
  __shared__ float4 s_nodes[NQ];
  const int tid = threadIdx.x;

  // Stage the 4 KB node table into LDS with the CDNA5 async path
  // (GVS addressing: vdst = LDS offset, vaddr = 32-bit offset, saddr = base).
  {
    const float* nbase = ws + OFF_NODES;
    unsigned goff    = (unsigned)tid * 16u;
    unsigned lds_off = (unsigned)(uintptr_t)&s_nodes[tid];
    asm volatile("global_load_async_to_lds_b128 %0, %1, %2 offset:0"
                 :
                 : "v"(lds_off), "v"(goff), "s"(nbase)
                 : "memory");
    asm volatile("s_wait_asynccnt 0" ::: "memory");
  }
  __syncthreads();

  const float S  = ws[OFF_S];
  const float kl = ws[OFF_KL];
  const _Float16* __restrict__ bp = (const _Float16*)(ws + OFF_BPACK);

  const int lane    = tid & 31;
  const int wave    = tid >> 5;
  const int rowBase = (blockIdx.x * 8 + wave) * 16;
  const int m       = lane & 15;   // A-matrix row this lane feeds
  const int hi      = lane >> 4;   // K-half selector

  const int* __restrict__ rowp = outM + (size_t)(rowBase + m) * NV;

  v8f acc = {};
  #pragma unroll
  for (int c = 0; c < 16; ++c) {
    // A in contiguous k order (same as B pack): lanes 0-15 carry K={0..15}, lanes 16-31
    // carry K={16..31} of this chunk.
    const int kb = c * 32 + hi * 16;
    const int4 p0 = *(const int4*)(rowp + kb);
    const int4 p1 = *(const int4*)(rowp + kb + 4);
    const int4 p2 = *(const int4*)(rowp + kb + 8);
    const int4 p3 = *(const int4*)(rowp + kb + 12);
    v16h a;
    a[0]  = (_Float16)p0.x; a[1]  = (_Float16)p0.y;
    a[2]  = (_Float16)p0.z; a[3]  = (_Float16)p0.w;
    a[4]  = (_Float16)p1.x; a[5]  = (_Float16)p1.y;
    a[6]  = (_Float16)p1.z; a[7]  = (_Float16)p1.w;
    a[8]  = (_Float16)p2.x; a[9]  = (_Float16)p2.y;
    a[10] = (_Float16)p2.z; a[11] = (_Float16)p2.w;
    a[12] = (_Float16)p3.x; a[13] = (_Float16)p3.y;
    a[14] = (_Float16)p3.z; a[15] = (_Float16)p3.w;
    const v16h b = *(const v16h*)(bp + ((c * 32 + lane) << 4));
    acc = __builtin_amdgcn_wmma_f32_16x16x32_f16(
        /*neg_a=*/false, a, /*neg_b=*/false, b,
        /*c_mod=*/(short)0, acc, /*reuse_a=*/false, /*reuse_b=*/false);
  }

  // Extract per-row dots from C fragment (VGPR r: lanes 0-15 -> M=r, N=lane;
  // lanes 16-31 -> M=8+r, N=lane-16).  Pairwise shfl_xor(1) combines the
  // hi/lo columns: lane lg+0 -> dotP = C[r,0]+C[r,1], lane lg+2 -> dotM.
  float dp = 0.0f, dm = 0.0f;
  #pragma unroll
  for (int r = 0; r < 16; ++r) {
    const float v   = acc[r & 7];
    const float tmp = v + __shfl_xor(v, 1, 32);
    const int   lg  = (r >> 3) << 4;
    const float pd  = __shfl(tmp, lg + 0, 32);
    const float md  = __shfl(tmp, lg + 2, 32);
    if (m == r) { dp = pd; dm = md; }
  }

  // a = correct, b = wrong; correct + wrong = S.
  const float yv  = (float)y[rowBase + m];
  const float aP  = 0.5f * (S + yv * dp);
  const float bP  = S - aP;
  const float am1 = aP - 1.0f;
  const float bm1 = bP - 1.0f;

  // Analytic upper bound on the log-integrand over (0, 0.5):
  // core(t) = am1*ln(t) + bm1*ln(1-t) peaks at t* = am1/(am1+bm1); logw <= 0,
  // so mx = core(t*) bounds every node -> single-pass logsumexp, no overflow.
  float tstar = am1 / fmaxf(am1 + bm1, 1e-3f);
  if (am1 <= 0.0f) tstar = 1e-5f;
  tstar = fminf(fmaxf(tstar, 1e-5f), 0.5f);
  const float mx = am1 * logf(tstar) + bm1 * log1pf(-tstar);

  // Lane pair (m, m+16) splits the 256 nodes.
  const int qb = hi << 7;
  float sum = 0.0f;
  #pragma unroll 4
  for (int q = 0; q < 128; ++q) {
    const float4 nd  = s_nodes[qb + q];           // broadcast ds_load_b128
    const float  lgq = __builtin_fmaf(am1, nd.x, __builtin_fmaf(bm1, nd.y, nd.z));
    sum += __expf(lgq - mx);
  }
  sum += __shfl_xor(sum, 16, 32);

  const float lse  = mx + __logf(sum);
  const float logB = lgammaf(aP) + lgammaf(bP) - lgammaf(aP + bP);
  float I = __expf(lse - logB);
  I = fminf(fmaxf(I, 0.0f), 1.0f);

  if (hi == 0) {
    const float4 o = make_float4(1.0f - 2.0f * I,
                                 0.5f * (1.0f - dm),
                                 0.5f * (1.0f + dm),
                                 kl);
    *(float4*)(dst + (size_t)(rowBase + m) * 4) = o;
  }
}

extern "C" void kernel_launch(void* const* d_in, const int* in_sizes, int n_in,
                              void* d_out, int out_size, void* d_ws, size_t ws_size,
                              hipStream_t stream) {
  const float* post_log = (const float*)d_in[0];   // [512]
  const float* prior    = (const float*)d_in[1];   // [512]
  const int*   outM     = (const int*)d_in[2];     // [B, 512]
  const int*   yv       = (const int*)d_in[3];     // [B, 1]
  float*       dst      = (float*)d_out;           // [B, 4]
  float*       ws       = (float*)d_ws;

  const int B = in_sizes[2] / NV;                  // 65536

  k_setup<<<1, 512, 0, stream>>>(post_log, prior, ws);
  k_main<<<B / 128, 256, 0, stream>>>(outM, yv, ws, dst);
}
